// Attention_41944650613314
// MI455X (gfx1250) — matrix operand order repaired and checked by reading, not compiled, not run on hardware
//
#include <hip/hip_runtime.h>
#include <hip/hip_bf16.h>

// ---------------------------------------------------------------------------
// Problem constants (from reference): B=64, L=1024, ENC=2048, DEC=512, ATT=512
// ---------------------------------------------------------------------------
#define B_   64
#define L_   1024
#define ENC_ 2048
#define DEC_ 512
#define ATT_ 512

typedef __bf16 bf16_t;
typedef __attribute__((ext_vector_type(8)))  __bf16 v8bf;
typedef __attribute__((ext_vector_type(16))) __bf16 v16bf;
typedef __attribute__((ext_vector_type(8)))  float  v8f;
typedef __attribute__((ext_vector_type(4)))  unsigned int v4u;
typedef __attribute__((ext_vector_type(4)))  int  v4i;
typedef __attribute__((ext_vector_type(8)))  int  v8i;

// ---------------------------------------------------------------------------
// Kernel 1: W_enc (ENC x ATT, f32) -> W_encT (ATT x ENC, bf16). L2-resident.
// ---------------------------------------------------------------------------
__global__ __launch_bounds__(256) void prep_wt(const float* __restrict__ W,
                                               bf16_t* __restrict__ WT) {
  __shared__ float tile[32][33];
  const int be = blockIdx.x;           // 64 tiles along ENC
  const int ba = blockIdx.y;           // 16 tiles along ATT
  const int tx = threadIdx.x & 31;
  const int ty = threadIdx.x >> 5;     // 0..7
#pragma unroll
  for (int i = 0; i < 4; ++i) {
    const int r = ty + i * 8;          // local e
    tile[r][tx] = W[(size_t)(be * 32 + r) * ATT_ + ba * 32 + tx];
  }
  __syncthreads();
#pragma unroll
  for (int i = 0; i < 4; ++i) {
    const int r = ty + i * 8;          // local a
    WT[(size_t)(ba * 32 + r) * ENC_ + be * 32 + tx] = (bf16_t)tile[tx][r];
  }
}

// ---------------------------------------------------------------------------
// Kernel 2: s2[b][n] = decoder_hidden[b]@W_dec[:,n] + b_dec[n] + b_enc[n]
// ---------------------------------------------------------------------------
__global__ __launch_bounds__(256) void prep_s2(const float* __restrict__ hid,
                                               const float* __restrict__ Wd,
                                               const float* __restrict__ bd,
                                               const float* __restrict__ ben,
                                               float* __restrict__ s2) {
  __shared__ float h[DEC_];
  const int b = blockIdx.x;
  const int t = threadIdx.x;
  for (int i = t; i < DEC_; i += 256) h[i] = hid[b * DEC_ + i];
  __syncthreads();
  for (int n = t; n < ATT_; n += 256) {
    float acc = 0.f;
#pragma unroll 8
    for (int k = 0; k < DEC_; ++k) acc += h[k] * Wd[(size_t)k * ATT_ + n];
    s2[b * ATT_ + n] = acc + bd[n] + ben[n];
  }
}

// ---------------------------------------------------------------------------
// TDM: one tensor_load_to_lds stages the whole B panel (512 rows x 32 bf16,
// i.e. W_encT[:, kb:kb+32]) into LDS, padded to an 80-byte row stride so the
// subsequent ds_load_b128 fragment reads are bank-conflict friendly.
// D# bitfields per CDNA5 ISA.
// ---------------------------------------------------------------------------
#define LDB 40  // bf16 elements per padded LDS row (80 bytes)

__device__ __forceinline__ void tdm_load_b_panel(unsigned int ldsOff,
                                                 const bf16_t* gsrc) {
  const unsigned long long ga = (unsigned long long)(uintptr_t)gsrc;
  v4u g0;
  g0[0] = 1u;                                   // count=1, user descriptor
  g0[1] = ldsOff;                               // lds_addr (bytes)
  g0[2] = (unsigned int)ga;                     // global_addr[31:0]
  g0[3] = (unsigned int)((ga >> 32) & 0x01FFFFFFu) | (2u << 30); // ga[56:32], type=2
  v8i g1;
  g1[0] = (1 << 16)      // data_size = 1 -> 2 bytes
        | (1 << 20)      // pad_enable
        | (3 << 22)      // pad_interval: 16 DWORDs (64B) between pads
        | (3 << 25);     // pad_amount: 4 DWORDs (16B) -> 80B row stride
  g1[1] = (int)(2048u << 16);  // tensor_dim0 = 2048 (bits 79:48, low half)
  g1[2] = (int)(512u << 16);   // tensor_dim0 hi=0 | tensor_dim1 = 512
  g1[3] = (int)(32u << 16);    // tensor_dim1 hi=0 | tile_dim0 = 32
  g1[4] = 512;                 // tile_dim1 = 512, tile_dim2 = 0
  g1[5] = 2048;                // tensor_dim0_stride (elements), low 32
  g1[6] = 0;                   // stride hi | tensor_dim1_stride lo
  g1[7] = 0;
  const v4i z4 = {0, 0, 0, 0};
#if __clang_major__ >= 23
  const v8i z8 = {0, 0, 0, 0, 0, 0, 0, 0};
  __builtin_amdgcn_tensor_load_to_lds(g0, g1, z4, z4, z8, 0);
#else
  __builtin_amdgcn_tensor_load_to_lds(g0, g1, z4, z4, 0);
#endif
}

// ---------------------------------------------------------------------------
// Kernel 3: fused  logits[b][l] = relu(enc@W_enc + s2[b]) @ W_full + b_full
// Block tile: M=64, N=512, K=2048. 8 waves x (4 M-subtiles x 4 N-tiles) = 16
// v_wmma_f32_16x16x32_bf16 per K-step per wave.
//   A: fp32 -> bf16 on the fly, double-buffered LDS (stride-40 pad).
//   B: TDM tensor_load_to_lds double-buffered panel, s_wait_tensorcnt synced.
// ---------------------------------------------------------------------------
#define LDA 40   // padded bf16 row stride (80B)

__global__ __launch_bounds__(256) void attn_gemm_logits(
    const float* __restrict__ enc,    // (B, L, ENC) f32
    const bf16_t* __restrict__ WT,    // (ATT, ENC)  bf16
    const float* __restrict__ s2,     // (B, ATT)
    const float* __restrict__ Wf,     // (ATT)
    const float* __restrict__ bfull,  // (1)
    float* __restrict__ logits)       // (B, L)
{
  __shared__ __attribute__((aligned(16))) bf16_t lB[2][512 * LDB]; // 2x40KB
  __shared__ __attribute__((aligned(16))) bf16_t lA[2][64 * LDA];  // 2x5KB
  __shared__ float lLog[64];

  const int tid  = threadIdx.x;
  const int lane = tid & 31;
  const int wave = tid >> 5;
  const int m0   = blockIdx.x * 64;     // first flattened (b*L + l) row
  const int b    = m0 >> 10;            // L = 1024; 64 | L so b is uniform

  if (tid < 64) lLog[tid] = 0.f;

  // --- A staging: thread -> 8 consecutive K of one of 64 rows ---
  const int srow = tid >> 2;            // 0..63
  const int sk   = (tid & 3) * 8;       // 0,8,16,24
  const float* encRow = enc + (size_t)(m0 + srow) * ENC_ + sk;

  // --- fragment lane mapping (CDNA5 16-bit A 16x32 / B 32x16 layouts) ---
  const int arow  = lane & 15;                 // M within subtile
  const int kbase = (lane < 16) ? 0 : 8;       // A: K chunks [0,8),[16,24)
  const int nlo   = lane & 15;                 // B/C: N within tile
  const int koff  = (lane < 16) ? 0 : 8;       // B: K chunks [0,8),[16,24)

  v8f acc[4][4];
#pragma unroll
  for (int mt = 0; mt < 4; ++mt)
#pragma unroll
    for (int nt = 0; nt < 4; ++nt) acc[mt][nt] = {};

  // --- prologue: stage K-slice 0 (A via VALU cvt, B via TDM) ---
  if (tid == 0) tdm_load_b_panel((unsigned int)(uintptr_t)&lB[0][0], WT + 0);
  {
    float4 f0 = *(const float4*)(encRow);
    float4 f1 = *(const float4*)(encRow + 4);
    v8bf o;
    o[0] = (bf16_t)f0.x; o[1] = (bf16_t)f0.y; o[2] = (bf16_t)f0.z; o[3] = (bf16_t)f0.w;
    o[4] = (bf16_t)f1.x; o[5] = (bf16_t)f1.y; o[6] = (bf16_t)f1.z; o[7] = (bf16_t)f1.w;
    *(v8bf*)&lA[0][srow * LDA + sk] = o;
  }
  if (wave == 0) __builtin_amdgcn_s_wait_tensorcnt(0);
  __syncthreads();

  for (int kb = 0; kb < ENC_; kb += 32) {
    const int cur = (kb >> 5) & 1;
    if (kb + 32 < ENC_) {   // stage next K-slice into the other buffers
      if (tid == 0)
        tdm_load_b_panel((unsigned int)(uintptr_t)&lB[cur ^ 1][0],
                         WT + (kb + 32));
      float4 f0 = *(const float4*)(encRow + kb + 32);
      float4 f1 = *(const float4*)(encRow + kb + 36);
      v8bf o;
      o[0] = (bf16_t)f0.x; o[1] = (bf16_t)f0.y; o[2] = (bf16_t)f0.z; o[3] = (bf16_t)f0.w;
      o[4] = (bf16_t)f1.x; o[5] = (bf16_t)f1.y; o[6] = (bf16_t)f1.z; o[7] = (bf16_t)f1.w;
      *(v8bf*)&lA[cur ^ 1][srow * LDA + sk] = o;
      __builtin_prefetch(encRow + kb + 64, 0, 1);   // global_prefetch_b8
    }

    // A fragments: 4 M-subtiles, two 16B LDS loads each (ISA 16-bit A layout)
    v16bf afrag[4];
#pragma unroll
    for (int mt = 0; mt < 4; ++mt) {
      const bf16_t* ap = &lA[cur][(mt * 16 + arow) * LDA + kbase];
      v8bf a0 = *(const v8bf*)(ap);
      v8bf a1 = *(const v8bf*)(ap + 16);
      afrag[mt] = __builtin_shufflevector(a0, a1, 0, 1, 2, 3, 4, 5, 6, 7,
                                          8, 9, 10, 11, 12, 13, 14, 15);
    }

    // 4 N-tiles: B fragment from TDM-staged LDS panel, reused by 4 M-subtiles
#pragma unroll
    for (int nt = 0; nt < 4; ++nt) {
      const int n = wave * 64 + nt * 16 + nlo;
      const bf16_t* bp = &lB[cur][n * LDB + koff];
      v8bf b0 = *(const v8bf*)(bp);
      v8bf b1 = *(const v8bf*)(bp + 16);
      v16bf bfrag = __builtin_shufflevector(b0, b1, 0, 1, 2, 3, 4, 5, 6, 7,
                                            8, 9, 10, 11, 12, 13, 14, 15);
#pragma unroll
      for (int mt = 0; mt < 4; ++mt)
        acc[mt][nt] = __builtin_amdgcn_wmma_f32_16x16x32_bf16(
            false, afrag[mt], false, bfrag, (short)0, acc[mt][nt], false, false);
    }

    if (wave == 0) __builtin_amdgcn_s_wait_tensorcnt(0); // next B panel landed
    __syncthreads();
  }

  // Epilogue: relu(c + s2[b][n]) * W_full[n], reduce n -> per-row logit.
  // C layout: lanes 0-15 -> (M=j, N=lane); lanes 16-31 -> (M=8+j, N=lane-16).
  const float* s2b = s2 + b * ATT_;
  float sv[4], wv[4];
#pragma unroll
  for (int nt = 0; nt < 4; ++nt) {
    const int n = wave * 64 + nt * 16 + nlo;
    sv[nt] = s2b[n];
    wv[nt] = Wf[n];
  }
  const int mbase = (lane < 16) ? 0 : 8;
#pragma unroll
  for (int mt = 0; mt < 4; ++mt) {
    float part[8] = {0.f, 0.f, 0.f, 0.f, 0.f, 0.f, 0.f, 0.f};
#pragma unroll
    for (int nt = 0; nt < 4; ++nt) {
#pragma unroll
      for (int j = 0; j < 8; ++j) {
        float r = acc[mt][nt][j] + sv[nt];
        r = fmaxf(r, 0.f);
        part[j] += r * wv[nt];
      }
    }
#pragma unroll
    for (int j = 0; j < 8; ++j)
      atomicAdd(&lLog[mt * 16 + mbase + j], part[j]);
  }
  __syncthreads();
  if (tid < 64) logits[m0 + tid] = lLog[tid] + bfull[0];
}

// ---------------------------------------------------------------------------
// Kernel 4: softmax over L per batch
// ---------------------------------------------------------------------------
__global__ __launch_bounds__(256) void attn_softmax(
    const float* __restrict__ logits, float* __restrict__ att) {
  __shared__ float red[256];
  const int b = blockIdx.x;
  const int t = threadIdx.x;
  const float* lg = logits + b * L_;
  float x0 = lg[t], x1 = lg[t + 256], x2 = lg[t + 512], x3 = lg[t + 768];
  float m = fmaxf(fmaxf(x0, x1), fmaxf(x2, x3));
  red[t] = m;
  __syncthreads();
  for (int s = 128; s > 0; s >>= 1) {
    if (t < s) red[t] = fmaxf(red[t], red[t + s]);
    __syncthreads();
  }
  const float mx = red[0];
  __syncthreads();
  float e0 = __expf(x0 - mx), e1 = __expf(x1 - mx);
  float e2 = __expf(x2 - mx), e3 = __expf(x3 - mx);
  red[t] = e0 + e1 + e2 + e3;
  __syncthreads();
  for (int s = 128; s > 0; s >>= 1) {
    if (t < s) red[t] += red[t + s];
    __syncthreads();
  }
  const float inv = 1.f / red[0];
  float* ob = att + b * L_;
  ob[t] = e0 * inv;
  ob[t + 256] = e1 * inv;
  ob[t + 512] = e2 * inv;
  ob[t + 768] = e3 * inv;
}

// ---------------------------------------------------------------------------
// Kernel 5: context[b][e] = sum_l att[b][l] * enc[b][l][e]  (streams 512 MB)
// ---------------------------------------------------------------------------
__global__ __launch_bounds__(256) void attn_context(
    const float* __restrict__ enc, const float* __restrict__ att,
    float* __restrict__ ctx) {
  __shared__ float sAtt[L_];
  const int chunk = blockIdx.x;   // 8 chunks of 256 along ENC
  const int b     = blockIdx.y;   // 64
  const int t     = threadIdx.x;
  for (int i = t; i < L_; i += 256) sAtt[i] = att[b * L_ + i];
  __syncthreads();
  const int e = chunk * 256 + t;
  const float* ep = enc + (size_t)b * L_ * ENC_ + e;
  float acc = 0.f;
#pragma unroll 4
  for (int l = 0; l < L_; ++l) acc += sAtt[l] * ep[(size_t)l * ENC_];
  ctx[b * ENC_ + e] = acc;
}

// ---------------------------------------------------------------------------
// Launch. d_out = [context (B*ENC f32) | att (B*L f32)]
// ws: [W_encT bf16 2MB | s2 128KB | logits 256KB]
// ---------------------------------------------------------------------------
extern "C" void kernel_launch(void* const* d_in, const int* in_sizes, int n_in,
                              void* d_out, int out_size, void* d_ws,
                              size_t ws_size, hipStream_t stream) {
  const float* enc  = (const float*)d_in[0];
  const float* hid  = (const float*)d_in[1];
  const float* Wenc = (const float*)d_in[2];
  const float* benc = (const float*)d_in[3];
  const float* Wdec = (const float*)d_in[4];
  const float* bdec = (const float*)d_in[5];
  const float* Wful = (const float*)d_in[6];
  const float* bful = (const float*)d_in[7];

  float* ctx = (float*)d_out;                 // B*ENC
  float* att = (float*)d_out + B_ * ENC_;     // B*L

  char* ws     = (char*)d_ws;
  bf16_t* WT   = (bf16_t*)ws;                                      // 2 MB
  float* s2    = (float*)(ws + 2u * 1024u * 1024u);                // 128 KB
  float* logit = (float*)(ws + 2u * 1024u * 1024u + 128u * 1024u); // 256 KB

  prep_wt<<<dim3(64, 16), 256, 0, stream>>>(Wenc, WT);
  prep_s2<<<dim3(B_), 256, 0, stream>>>(hid, Wdec, bdec, benc, s2);
  attn_gemm_logits<<<dim3((B_ * L_) / 64), 256, 0, stream>>>(enc, WT, s2, Wful,
                                                             bful, logit);
  attn_softmax<<<dim3(B_), 256, 0, stream>>>(logit, att);
  attn_context<<<dim3(ENC_ / 256, B_), 256, 0, stream>>>(enc, att, ctx);
}
